// RobertaSelfAttention_82420422410299
// MI455X (gfx1250) — matrix operand order repaired and checked by reading, not compiled, not run on hardware
//
#include <hip/hip_runtime.h>

// ---------------------------------------------------------------------------
// RobertaSelfAttention forward for MI455X (gfx1250, wave32, WMMA bf16).
//   N=4, L=2048, E=1024, H=16, D=64
// 3x bf16 WMMA projection GEMMs (double-buffered LDS, register prefetch)
// -> flash attention (double-buffered K/V tiles staged with async
// global->LDS, online softmax in log2 domain, scores never hit HBM).
// f32 accumulation everywhere.
// ---------------------------------------------------------------------------

typedef __attribute__((ext_vector_type(16))) __bf16 v16bf;
typedef __attribute__((ext_vector_type(8)))  __bf16 v8bf;
typedef __attribute__((ext_vector_type(8)))  float  v8f;
typedef __attribute__((ext_vector_type(4)))  float  v4f;
typedef __attribute__((ext_vector_type(4)))  int    v4i;

#define E_DIM 1024
#define H_NUM 16
#define D_DIM 64
#define N_B   4
#define L_SEQ 2048
#define NL    (N_B * L_SEQ)
#define LOG2E 1.4426950408889634f

// Async global->LDS (gfx1250: GLOBAL_LOAD_ASYNC_TO_LDS_B128, ASYNCcnt).
// Prototype (from hipcc diagnostic): (global int4*, local int4*, imm, imm).
#if defined(__has_builtin)
#  if __has_builtin(__builtin_amdgcn_global_load_async_to_lds_b128)
#    define USE_ASYNC_LDS 1
#  endif
#endif
#ifndef USE_ASYNC_LDS
#  define USE_ASYNC_LDS 0
#endif

static __device__ inline void cp_b128(__bf16* dst_lds, const __bf16* src) {
#if USE_ASYNC_LDS
  __builtin_amdgcn_global_load_async_to_lds_b128(
      (__attribute__((address_space(1))) v4i*)(void*)src,
      (__attribute__((address_space(3))) v4i*)(void*)dst_lds, 0, 0);
#else
  *(v8bf*)dst_lds = *(const v8bf*)src;
#endif
}

static __device__ inline void async_join() {
#if USE_ASYNC_LDS
#  if __has_builtin(__builtin_amdgcn_s_wait_asynccnt)
  __builtin_amdgcn_s_wait_asynccnt(0);
#  else
  asm volatile("s_wait_asynccnt 0" ::: "memory");
#  endif
#endif
}

static __device__ inline v16bf cat16(v8bf lo, v8bf hi) {
  v16bf r;
#pragma unroll
  for (int i = 0; i < 8; i++) { r[i] = lo[i]; r[i + 8] = hi[i]; }
  return r;
}

static __device__ inline v8f v8f_zero() {
  v8f z;
#pragma unroll
  for (int i = 0; i < 8; i++) z[i] = 0.f;
  return z;
}

// ---------------------------------------------------------------------------
// Projection: out = (x @ W^T + b) * scale -> bf16, head-major layout.
// VLAYOUT 0 -> [N,H,L,D] (Q,K); VLAYOUT 1 -> [N,H,D,L] (V transposed so the
// attention P*V B-fragments are contiguous in the key dimension).
// 256 threads (8 waves), 128x128 tile, K-step 32, double-buffered LDS with
// register prefetch. Wave (wy,wx) computes a 32x64 sub-tile (8 C fragments).
// ---------------------------------------------------------------------------
template <int VLAYOUT>
__global__ __launch_bounds__(256) void proj_kernel(
    const float* __restrict__ x, const float* __restrict__ W,
    const float* __restrict__ bias, __bf16* __restrict__ out, float scale)
{
  __shared__ __bf16 Alds[2][128 * 32];   // [m][k]  2 x 8 KB
  __shared__ __bf16 Blds[2][128 * 32];   // [n][k]  2 x 8 KB (W row-major = B^T)

  const int tid  = threadIdx.x;
  const int lane = tid & 31;
  const int w    = tid >> 5;
  const int wy   = w >> 1;            // 0..3 -> 32-row strip
  const int wx   = w & 1;             // 0..1 -> 64-col strip
  const int mrow = lane & 15;
  const int hi   = lane >> 4;
  const int col0 = blockIdx.x * 128;
  const int row0 = blockIdx.y * 128;

  // Staging: each thread owns one half-row (16 fp32 -> 16 bf16).
  const int sm  = tid >> 1;           // 0..127
  const int skb = (tid & 1) * 16;
  const float* aSrc = x + (size_t)(row0 + sm) * E_DIM + skb;
  const float* bSrc = W + (size_t)(col0 + sm) * E_DIM + skb;

  v4f ar[4], br[4];
  auto ldreg = [&](const float* p, v4f* r) {
#pragma unroll
    for (int j = 0; j < 4; j++) r[j] = *(const v4f*)(p + j * 4);
  };
  auto stlds = [&](__bf16* base, const v4f* r) {
    v8bf b0, b1;
#pragma unroll
    for (int i = 0; i < 4; i++) {
      b0[i] = (__bf16)r[0][i]; b0[i + 4] = (__bf16)r[1][i];
      b1[i] = (__bf16)r[2][i]; b1[i + 4] = (__bf16)r[3][i];
    }
    *(v8bf*)(base + sm * 32 + skb)     = b0;
    *(v8bf*)(base + sm * 32 + skb + 8) = b1;
  };

  ldreg(aSrc, ar);
  ldreg(bSrc, br);
  stlds(Alds[0], ar);
  stlds(Blds[0], br);
  __syncthreads();

  v8f acc[2][4];
#pragma unroll
  for (int mf = 0; mf < 2; mf++)
#pragma unroll
    for (int nf = 0; nf < 4; nf++) acc[mf][nf] = v8f_zero();

  const int NSTEP = E_DIM / 32;
  for (int step = 0; step < NSTEP; ++step) {
    const int cur = step & 1;
    if (step + 1 < NSTEP) {               // prefetch next K-slab into regs
      ldreg(aSrc + (step + 1) * 32, ar);
      ldreg(bSrc + (step + 1) * 32, br);
    }

    // A fragments (16x32 bf16): element i -> k = (i<8 ? i : i+8) + 8*hi
    v16bf afrag[2];
#pragma unroll
    for (int mf = 0; mf < 2; mf++) {
      const __bf16* ap = &Alds[cur][(wy * 32 + mf * 16 + mrow) * 32 + hi * 8];
      afrag[mf] = cat16(*(const v8bf*)ap, *(const v8bf*)(ap + 16));
    }
#pragma unroll
    for (int nf = 0; nf < 4; nf++) {
      // B fragment (32x16 bf16): element i -> k = (i<8 ? i : i+8) + 8*hi, n = mrow
      const __bf16* bp = &Blds[cur][(wx * 64 + nf * 16 + mrow) * 32 + hi * 8];
      v16bf bfrag = cat16(*(const v8bf*)bp, *(const v8bf*)(bp + 16));
#pragma unroll
      for (int mf = 0; mf < 2; mf++)
        acc[mf][nf] = __builtin_amdgcn_wmma_f32_16x16x32_bf16(
            false, afrag[mf], false, bfrag, (short)0, acc[mf][nf], false, false);
    }

    if (step + 1 < NSTEP) {               // store prefetched slab to other buf
      stlds(Alds[cur ^ 1], ar);
      stlds(Blds[cur ^ 1], br);
    }
    __syncthreads();
  }

  // Epilogue: bias + scale, scatter bf16 into head-major layout.
#pragma unroll
  for (int mf = 0; mf < 2; mf++) {
#pragma unroll
    for (int nf = 0; nf < 4; nf++) {
      const int gc = col0 + wx * 64 + nf * 16 + mrow;
      const float bv = bias[gc];
      const int hh = gc >> 6;
      const int dd = gc & 63;
#pragma unroll
      for (int r = 0; r < 8; r++) {
        const int gr = row0 + wy * 32 + mf * 16 + r + 8 * hi;
        const int nb = gr >> 11;
        const int l  = gr & 2047;
        const float val = (acc[mf][nf][r] + bv) * scale;
        size_t off;
        if (VLAYOUT == 0)
          off = ((size_t)((nb * H_NUM + hh) * L_SEQ + l)) * D_DIM + dd;
        else
          off = ((size_t)((nb * H_NUM + hh) * D_DIM + dd)) * L_SEQ + l;
        out[off] = (__bf16)val;
      }
    }
  }
}

// ---------------------------------------------------------------------------
// Flash attention. 256 threads (8 waves = 128 queries) per (n,h,q-tile).
// Double-buffered 64-key K/V tiles staged via async global->LDS (overlaps
// with WMMA + softmax of the current tile). Scores are pre-scaled by log2e
// (folded into Q and the mask) so softmax uses raw v_exp_f32 (exp2f).
// ---------------------------------------------------------------------------
__global__ __launch_bounds__(256) void attn_kernel(
    const __bf16* __restrict__ Qb, const __bf16* __restrict__ Kb,
    const __bf16* __restrict__ Vb, const float* __restrict__ mask,
    float* __restrict__ out)
{
  __shared__ __bf16 Kbuf[2][64 * 64];     // [key][d]   2 x 8 KB
  __shared__ __bf16 Vbuf[2][64 * 64];     // [d][key]   2 x 8 KB
  __shared__ __bf16 Plds[8][16 * 64];     // per-wave P   16 KB
  __shared__ float  Mbuf[2][64];

  const int tid  = threadIdx.x;
  const int lane = tid & 31;
  const int w    = tid >> 5;
  const int mrow = lane & 15;
  const int hi   = lane >> 4;

  const int qt = blockIdx.x & 15;               // L/128 = 16 query tiles
  const int h  = (blockIdx.x >> 4) & 15;
  const int nb = blockIdx.x >> 8;
  const int q0 = qt * 128 + w * 16;

  const size_t headQK = (size_t)(nb * H_NUM + h) * L_SEQ * D_DIM; // [N,H,L,D]
  const size_t headV  = (size_t)(nb * H_NUM + h) * D_DIM * L_SEQ; // [N,H,D,L]

  // Staging: each thread owns a 16-element chunk of K and V tiles.
  const int srow = tid >> 2;          // 0..63
  const int scol = (tid & 3) * 16;    // 0,16,32,48

  auto stage = [&](int b, int kb) {
    const __bf16* ks = Kb + headQK + (size_t)(kb + srow) * D_DIM + scol;
    cp_b128(&Kbuf[b][srow * 64 + scol],     ks);
    cp_b128(&Kbuf[b][srow * 64 + scol + 8], ks + 8);
    const __bf16* vs = Vb + headV + (size_t)srow * L_SEQ + kb + scol;
    cp_b128(&Vbuf[b][srow * 64 + scol],     vs);
    cp_b128(&Vbuf[b][srow * 64 + scol + 8], vs + 8);
    if (tid < 64) Mbuf[b][tid] = mask[nb * L_SEQ + kb + tid] * LOG2E;
  };

  // Q fragments (2 d-chunks of 32) straight from global (contiguous in d).
  // Q was pre-scaled by (1/sqrt(D))*log2e in the projection.
  v16bf qfrag[2];
#pragma unroll
  for (int c = 0; c < 2; c++) {
    const __bf16* qp =
        Qb + headQK + (size_t)(q0 + mrow) * D_DIM + c * 32 + hi * 8;
    qfrag[c] = cat16(*(const v8bf*)qp, *(const v8bf*)(qp + 16));
  }

  v8f o[4];
#pragma unroll
  for (int nf = 0; nf < 4; nf++) o[nf] = v8f_zero();
  float mr[8], lr[8];
#pragma unroll
  for (int r = 0; r < 8; r++) { mr[r] = -1e30f; lr[r] = 0.f; }

  stage(0, 0);
  async_join();
  __syncthreads();

  const int NTILE = L_SEQ / 64;
  for (int step = 0; step < NTILE; ++step) {
    const int cur = step & 1;
    if (step + 1 < NTILE) stage(cur ^ 1, (step + 1) * 64);  // overlapped

    // S = Q K^T (log2-domain) : 4 fragments of 16 keys, 2 d-chunks each.
    v8f s[4];
#pragma unroll
    for (int nf = 0; nf < 4; nf++) {
      v8f sf = v8f_zero();
#pragma unroll
      for (int c = 0; c < 2; c++) {
        const __bf16* kp = &Kbuf[cur][(nf * 16 + mrow) * 64 + c * 32 + hi * 8];
        v16bf bfrag = cat16(*(const v8bf*)kp, *(const v8bf*)(kp + 16));
        sf = __builtin_amdgcn_wmma_f32_16x16x32_bf16(
            false, qfrag[c], false, bfrag, (short)0, sf, false, false);
      }
      const float mv = Mbuf[cur][nf * 16 + mrow];
#pragma unroll
      for (int r = 0; r < 8; r++) sf[r] += mv;
      s[nf] = sf;
    }

    // Online softmax per row (row spread over 16 lanes of a half-wave).
    // exp2f -> single v_exp_f32 (base-2 exponential on AMD).
#pragma unroll
    for (int r = 0; r < 8; r++) {
      float tmax = s[0][r];
#pragma unroll
      for (int nf = 1; nf < 4; nf++) tmax = fmaxf(tmax, s[nf][r]);
#pragma unroll
      for (int off = 1; off < 16; off <<= 1)
        tmax = fmaxf(tmax, __shfl_xor(tmax, off, 16));
      const float nm    = fmaxf(mr[r], tmax);
      const float alpha = exp2f(mr[r] - nm);
      mr[r] = nm;
      float rsum = 0.f;
#pragma unroll
      for (int nf = 0; nf < 4; nf++) {
        const float p = exp2f(s[nf][r] - nm);
        s[nf][r] = p;
        rsum += p;
      }
#pragma unroll
      for (int off = 1; off < 16; off <<= 1)
        rsum += __shfl_xor(rsum, off, 16);
      lr[r] = lr[r] * alpha + rsum;
#pragma unroll
      for (int nf = 0; nf < 4; nf++) o[nf][r] *= alpha;
    }

    // Re-layout P (C-fragment f32) -> A-fragment bf16 via per-wave LDS.
#pragma unroll
    for (int nf = 0; nf < 4; nf++)
#pragma unroll
      for (int r = 0; r < 8; r++)
        Plds[w][(r + 8 * hi) * 64 + nf * 16 + mrow] = (__bf16)s[nf][r];

    v16bf pfrag[2];
#pragma unroll
    for (int c = 0; c < 2; c++) {
      const __bf16* pp = &Plds[w][mrow * 64 + c * 32 + hi * 8];
      pfrag[c] = cat16(*(const v8bf*)pp, *(const v8bf*)(pp + 16));
    }

    // O += P V
#pragma unroll
    for (int nf = 0; nf < 4; nf++) {
#pragma unroll
      for (int c = 0; c < 2; c++) {
        const __bf16* vp = &Vbuf[cur][(nf * 16 + mrow) * 64 + c * 32 + hi * 8];
        v16bf vfrag = cat16(*(const v8bf*)vp, *(const v8bf*)(vp + 16));
        o[nf] = __builtin_amdgcn_wmma_f32_16x16x32_bf16(
            false, pfrag[c], false, vfrag, (short)0, o[nf], false, false);
      }
    }

    async_join();      // next-tile async copies must land before the barrier
    __syncthreads();
  }

  // Epilogue: O * (1/l), write fp32 [N, L, E]. One divide per row.
  float linv[8];
#pragma unroll
  for (int r = 0; r < 8; r++) linv[r] = 1.0f / lr[r];
#pragma unroll
  for (int nf = 0; nf < 4; nf++) {
    const int d = nf * 16 + mrow;
#pragma unroll
    for (int r = 0; r < 8; r++) {
      const int q = q0 + r + 8 * hi;
      out[((size_t)(nb * L_SEQ + q)) * E_DIM + h * D_DIM + d] =
          o[nf][r] * linv[r];
    }
  }
}

// ---------------------------------------------------------------------------
// Launch. Inputs: x, attn_mask, Wq, bq, Wk, bk, Wv, bv. Output fp32 [N,L,E].
// Workspace: Q[N,H,L,D] | K[N,H,L,D] | V[N,H,D,L], bf16 (48 MB total).
// ---------------------------------------------------------------------------
extern "C" void kernel_launch(void* const* d_in, const int* in_sizes, int n_in,
                              void* d_out, int out_size, void* d_ws, size_t ws_size,
                              hipStream_t stream) {
  (void)in_sizes; (void)n_in; (void)out_size; (void)ws_size;
  const float* x    = (const float*)d_in[0];
  const float* mask = (const float*)d_in[1];
  const float* Wq   = (const float*)d_in[2];
  const float* bq   = (const float*)d_in[3];
  const float* Wk   = (const float*)d_in[4];
  const float* bk   = (const float*)d_in[5];
  const float* Wv   = (const float*)d_in[6];
  const float* bv   = (const float*)d_in[7];
  float* out = (float*)d_out;

  const size_t elems = (size_t)N_B * H_NUM * L_SEQ * D_DIM;  // 8M
  __bf16* Qb = (__bf16*)d_ws;
  __bf16* Kb = Qb + elems;
  __bf16* Vb = Kb + elems;

  dim3 pgrid(E_DIM / 128, NL / 128);   // 8 x 64
  const float qscale = 0.125f * LOG2E; // (1/sqrt(64)) * log2(e)
  proj_kernel<0><<<pgrid, 256, 0, stream>>>(x, Wq, bq, Qb, qscale);
  proj_kernel<0><<<pgrid, 256, 0, stream>>>(x, Wk, bk, Kb, 1.0f);
  proj_kernel<1><<<pgrid, 256, 0, stream>>>(x, Wv, bv, Vb, 1.0f);

  const int ablocks = N_B * H_NUM * (L_SEQ / 128);  // 1024
  attn_kernel<<<ablocks, 256, 0, stream>>>(Qb, Kb, Vb, mask, out);
}
